// EfficientHybridAttentionBlock_43705587204173
// MI455X (gfx1250) — matrix operand order repaired and checked by reading, not compiled, not run on hardware
//
#include <hip/hip_runtime.h>
#include <hip/hip_bf16.h>
#include <math.h>

#define H_      56
#define W_      56
#define L_      (H_*W_)            // 3136
#define C_      128
#define WSZ     7
#define SHIFT_  3
#define HEADS_  4
#define HD_     32
#define NWIN    64
#define B_      64
#define NTOK    (B_*L_)            // 200704
#define MLP_    512
#define EPS_    1e-5f

typedef __attribute__((ext_vector_type(16))) _Float16 v16h;
typedef __attribute__((ext_vector_type(8)))  float    v8f;

// ---------------------------------------------------------------- WMMA helpers
__device__ __forceinline__ v8f wmma16(v16h a, v16h b, v8f c) {
  // D = A(16x32 f16) * B(32x16 f16) + C(16x16 f32)
  return __builtin_amdgcn_wmma_f32_16x16x32_f16(false, a, false, b, (short)0, c,
                                                false, false);
}

// A fragment 16x32 f16 from row-major [M][K] storage (ld in halves).
// k order used here (B loaders match): lanes 0-15 -> M=lane, K halves {0..15};
//          lanes 16-31 -> M=lane-16, K halves {16..31}
// Two contiguous 16B runs per lane -> ds_load_b128 x2.
__device__ __forceinline__ v16h frag_a(const _Float16* base, int ld) {
  const int lane = threadIdx.x & 31;
  const int row  = lane & 15;
  const int k0   = (lane >> 4) << 4;        // 0 or 16
  const _Float16* p = base + row * ld;
  v16h a;
#pragma unroll
  for (int i = 0; i < 8; ++i) a[i]     = p[k0 + i];
#pragma unroll
  for (int i = 0; i < 8; ++i) a[8 + i] = p[8 + k0 + i];
  return a;
}

// B fragment 32x16 where the source is row-major [N][K] (i.e. B^T storage);
// 16 contiguous halves per lane -> ds_load_b128 x2.
__device__ __forceinline__ v16h frag_b_tr(const _Float16* base, int ld) {
  const int lane = threadIdx.x & 31;
  const int col  = lane & 15;
  const int k0   = (lane >> 4) << 4;        // 0 or 16
  return *(const v16h*)(base + col * ld + k0);
}

// B fragment from pre-swizzled weight storage: one contiguous 32B record per
// lane (fully coalesced 1KB per wave).
__device__ __forceinline__ v16h frag_b_sw(const _Float16* tile_base) {
  const int lane = threadIdx.x & 31;
  return *(const v16h*)(tile_base + lane * 16);
}

__device__ __forceinline__ float wave_sum(float v) {
#pragma unroll
  for (int m = 16; m >= 1; m >>= 1) v += __shfl_xor(v, m, 32);
  return v;
}

// ---------------------------------------------------------------- small utils
__global__ void k_zero(float* p, int n) {
  int i = blockIdx.x * 256 + threadIdx.x;
  if (i < n) p[i] = 0.f;
}

// Swizzle f32 weight W[K][N] into per-lane-contiguous WMMA B fragments (f16).
// Record layout: ((nt*ktiles + kt)*32 + lane)*16 + i, lane: col=lane&15,
// k0=(lane>>4)*16, value = W[kt*32 + k0 + i][nt*16 + col].
__global__ void k_swz(const float* __restrict__ src, _Float16* __restrict__ dst,
                      int K, int N) {
  const int ktiles = K >> 5;
  const int total = (K >> 5) * (N >> 4) * 32;
  int t = blockIdx.x * 256 + threadIdx.x;
  if (t >= total) return;
  const int lane = t & 31, tile = t >> 5;
  const int kt = tile % ktiles, nt = tile / ktiles;
  const int col = lane & 15, k0 = (lane >> 4) << 4;
  _Float16 rec[16];
#pragma unroll
  for (int i = 0; i < 16; ++i)
    rec[i] = (_Float16)src[(size_t)(kt * 32 + k0 + i) * N + nt * 16 + col];
  *(v16h*)(dst + (size_t)t * 16) = *(const v16h*)rec;
}

// ------------------------------------------------- LN1 + ECA channel-sum pass
__global__ void __launch_bounds__(256) k_ln1(
    const float* __restrict__ x, const float* __restrict__ g,
    const float* __restrict__ bb, _Float16* __restrict__ xsh,
    float* __restrict__ sums) {
  __shared__ float bs[C_];
  const int tid = threadIdx.x, wave = tid >> 5, lane = tid & 31;
  const size_t tok = (size_t)blockIdx.x * 8 + wave;
  const int bi = (int)(tok / L_);
  if (tid < C_) bs[tid] = 0.f;
  __syncthreads();

  const int c0 = lane * 4;
  float4 v = *(const float4*)(x + tok * C_ + c0);
  float mu = wave_sum(v.x + v.y + v.z + v.w) * (1.f / 128.f);
  float d0 = v.x - mu, d1 = v.y - mu, d2 = v.z - mu, d3 = v.w - mu;
  float var = wave_sum(d0*d0 + d1*d1 + d2*d2 + d3*d3) * (1.f / 128.f);
  float rs = rsqrtf(var + EPS_);
  float o0 = d0 * rs * g[c0+0] + bb[c0+0];
  float o1 = d1 * rs * g[c0+1] + bb[c0+1];
  float o2 = d2 * rs * g[c0+2] + bb[c0+2];
  float o3 = d3 * rs * g[c0+3] + bb[c0+3];
  _Float16 hh[4] = {(_Float16)o0, (_Float16)o1, (_Float16)o2, (_Float16)o3};
  *(float2*)(xsh + tok * C_ + c0) = *(const float2*)hh;

  atomicAdd(&bs[c0+0], o0); atomicAdd(&bs[c0+1], o1);
  atomicAdd(&bs[c0+2], o2); atomicAdd(&bs[c0+3], o3);
  __syncthreads();
  if (tid < C_) atomicAdd(&sums[bi * C_ + tid], bs[tid]);
}

// ECA gate: only the center tap of the 'SAME' conv over a length-1 axis survives.
__global__ void k_gate(const float* __restrict__ sums,
                       const float* __restrict__ ecaw, float* __restrict__ gate) {
  __shared__ float red[C_];
  const int b = blockIdx.x, c = threadIdx.x;
  red[c] = sums[b * C_ + c] * (1.f / (float)L_) * ecaw[C_ + c];  // eca_w[1][c][0]
  __syncthreads();
  for (int s = 64; s > 0; s >>= 1) {
    if (c < s) red[c] += red[c + s];
    __syncthreads();
  }
  if (c == 0) gate[b] = 1.f / (1.f + __expf(-red[0]));
}

// ------------------------------------------------- shifted window attention
__global__ void __launch_bounds__(256) k_attn(
    const _Float16* __restrict__ xsh, const _Float16* __restrict__ qkvws,
    const float* __restrict__ qkvb, const _Float16* __restrict__ projws,
    const float* __restrict__ projb, const float* __restrict__ relbias,
    const int* __restrict__ relidx, const float* __restrict__ amask,
    float* __restrict__ aout) {
  __shared__ _Float16 sXw[64 * 128];   // window tokens (rows >=49 zero)   16KB
  __shared__ _Float16 sQK[64 * 64];    // per-head Q(0..31)|K(32..63)       8KB
  __shared__ _Float16 sVt[32 * 64];    // per-head V transposed [dim][tok]  4KB
  __shared__ float    sS[64 * 64];     // scores (P f16 aliases rows)      16KB
  __shared__ _Float16 sO[64 * 128];    // per-window attention output      16KB

  const int tid = threadIdx.x;
  const int lane = tid & 31, wave = tid >> 5;
  const int bi = blockIdx.x >> 6, wi = blockIdx.x & 63;
  const int wh = wi >> 3, ww = wi & 7;

  // stage (cyclic shift folded into index: orig = (shifted + SHIFT) % 56)
#pragma unroll
  for (int i = 0; i < 4; ++i) {
    int ch = tid + 256 * i;                       // 1024 chunks of 8 halves
    int row = ch >> 4, c0 = (ch & 15) << 3;
    float4 vv = {0.f, 0.f, 0.f, 0.f};
    if (row < 49) {
      int lh = row / 7, lw = row % 7;
      int h0 = (wh * 7 + lh + SHIFT_) % H_;
      int w0 = (ww * 7 + lw + SHIFT_) % W_;
      int tok = h0 * W_ + w0;
      vv = *(const float4*)(xsh + ((size_t)bi * L_ + tok) * C_ + c0);
    }
    ((float4*)sXw)[ch] = vv;
  }
  __syncthreads();

  const float scale = 0.17677669529663687f;      // 1/sqrt(32)

  for (int h = 0; h < HEADS_; ++h) {
    // ---- per-head QKV: [64x128] @ [128x96] (cols picked out of qkv_w)
    for (int t = wave; t < 24; t += 8) {
      int m = t / 6, n = t % 6;
      int seg = n >> 1;                           // 0=Q 1=K 2=V
      int gcol = seg * C_ + h * HD_ + (n & 1) * 16;
      int nt = gcol >> 4;                         // global N-tile, ktiles=4
      v8f acc = {};
#pragma unroll
      for (int kt = 0; kt < 4; ++kt) {
        v16h a = frag_a(sXw + m * 16 * C_ + kt * 32, C_);
        v16h b = frag_b_sw(qkvws + ((size_t)(nt * 4 + kt) << 9));
        acc = wmma16(a, b, acc);
      }
      int col = lane & 15, r0 = (lane >> 4) << 3;
      float bias = qkvb[gcol + col];
      if (seg < 2) {                              // Q|K token-major
#pragma unroll
        for (int i = 0; i < 8; ++i)
          sQK[(m * 16 + r0 + i) * 64 + n * 16 + col] = (_Float16)(acc[i] + bias);
      } else {                                    // V dim-major (transposed)
        int dloc = (n & 1) * 16 + col;
#pragma unroll
        for (int i = 0; i < 8; ++i)
          sVt[dloc * 64 + m * 16 + r0 + i] = (_Float16)(acc[i] + bias);
      }
    }
    __syncthreads();

    // ---- S = Q @ K^T  (K token-major -> transposed-B loader, contiguous)
    for (int t = wave; t < 16; t += 8) {
      int m = t >> 2, n = t & 3;
      v16h a = frag_a(sQK + m * 16 * 64, 64);
      v16h b = frag_b_tr(sQK + n * 16 * 64 + 32, 64);
      v8f acc = {};
      acc = wmma16(a, b, acc);
      int col = lane & 15, r0 = (lane >> 4) << 3;
#pragma unroll
      for (int i = 0; i < 8; ++i)
        sS[(m * 16 + r0 + i) * 64 + n * 16 + col] = acc[i];
    }
    __syncthreads();

    // ---- scale + relative bias + shift mask + pad columns
    for (int idx = tid; idx < 4096; idx += 256) {
      int q = idx >> 6, k = idx & 63;
      float s = sS[idx] * scale;
      if (k >= 49) s = -1e30f;
      else if (q < 49)
        s += relbias[relidx[q * 49 + k] * HEADS_ + h] +
             amask[((size_t)wi * 49 + q) * 49 + k];
      sS[idx] = s;
    }
    __syncthreads();

    // ---- row softmax -> fp16 P aliased onto the f32 score rows
    if (tid < 64) {
      float* row = sS + tid * 64;
      float mx = -1e30f;
      for (int k = 0; k < 64; ++k) mx = fmaxf(mx, row[k]);
      float sum = 0.f;
      for (int k = 0; k < 64; ++k) sum += __expf(row[k] - mx);
      float inv = 1.f / sum;
      _Float16* P = reinterpret_cast<_Float16*>(row);
      for (int k = 0; k < 64; ++k) {
        float e = __expf(row[k] - mx) * inv;   // read f32[k] before writing h[k]
        P[k] = (_Float16)e;
      }
    }
    __syncthreads();

    // ---- O_h = P @ V  (V stored dim-major -> contiguous B frags)
    {
      int m = wave >> 1, n = wave & 1;
      v8f acc = {};
#pragma unroll
      for (int kt = 0; kt < 2; ++kt) {
        v16h a = frag_a(reinterpret_cast<const _Float16*>(sS) + m * 16 * 128 + kt * 32, 128);
        v16h b = frag_b_tr(sVt + (n * 16) * 64 + kt * 32, 64);
        acc = wmma16(a, b, acc);
      }
      int col = lane & 15, r0 = (lane >> 4) << 3;
#pragma unroll
      for (int i = 0; i < 8; ++i)
        sO[(m * 16 + r0 + i) * C_ + h * HD_ + n * 16 + col] = (_Float16)acc[i];
    }
    __syncthreads();
  }

  // ---- proj: [64x128] @ [128x128], reverse shift folded into the store
  for (int t = wave; t < 32; t += 8) {
    int m = t >> 3, n = t & 7;
    v8f acc = {};
#pragma unroll
    for (int kt = 0; kt < 4; ++kt) {
      v16h a = frag_a(sO + m * 16 * C_ + kt * 32, C_);
      v16h b = frag_b_sw(projws + ((size_t)(n * 4 + kt) << 9));
      acc = wmma16(a, b, acc);
    }
    int col = lane & 15, r0 = (lane >> 4) << 3;
    float bias = projb[n * 16 + col];
#pragma unroll
    for (int i = 0; i < 8; ++i) {
      int p = m * 16 + r0 + i;
      if (p < 49) {
        int lh = p / 7, lw = p % 7;
        int h0 = (wh * 7 + lh + SHIFT_) % H_;
        int w0 = (ww * 7 + lw + SHIFT_) % W_;
        int tok = h0 * W_ + w0;
        aout[((size_t)bi * L_ + tok) * C_ + n * 16 + col] = acc[i] + bias;
      }
    }
  }
}

// ---------------- residual + cattn + LN2
__global__ void __launch_bounds__(256) k_mid(
    const float* __restrict__ x, const float* __restrict__ aout,
    const float* __restrict__ gate,
    const float* __restrict__ g1, const float* __restrict__ b1,
    const float* __restrict__ g2, const float* __restrict__ b2,
    float* __restrict__ xmid, _Float16* __restrict__ xnh) {
  const int tid = threadIdx.x, wave = tid >> 5, lane = tid & 31;
  const size_t tok = (size_t)blockIdx.x * 8 + wave;
  const int bi = (int)(tok / L_);
  const int c0 = lane * 4;

  float4 xv = *(const float4*)(x + tok * C_ + c0);
  float4 av = *(const float4*)(aout + tok * C_ + c0);

  float mu = wave_sum(xv.x + xv.y + xv.z + xv.w) * (1.f / 128.f);
  float d0 = xv.x - mu, d1 = xv.y - mu, d2 = xv.z - mu, d3 = xv.w - mu;
  float var = wave_sum(d0*d0 + d1*d1 + d2*d2 + d3*d3) * (1.f / 128.f);
  float rs = rsqrtf(var + EPS_);
  float gg = gate[bi];

  float m0 = xv.x + av.x + gg * (d0 * rs * g1[c0+0] + b1[c0+0]);
  float m1 = xv.y + av.y + gg * (d1 * rs * g1[c0+1] + b1[c0+1]);
  float m2 = xv.z + av.z + gg * (d2 * rs * g1[c0+2] + b1[c0+2]);
  float m3 = xv.w + av.w + gg * (d3 * rs * g1[c0+3] + b1[c0+3]);

  float mu2 = wave_sum(m0 + m1 + m2 + m3) * (1.f / 128.f);
  float e0 = m0 - mu2, e1 = m1 - mu2, e2 = m2 - mu2, e3 = m3 - mu2;
  float var2 = wave_sum(e0*e0 + e1*e1 + e2*e2 + e3*e3) * (1.f / 128.f);
  float rs2 = rsqrtf(var2 + EPS_);

  float4 mv = {m0, m1, m2, m3};
  *(float4*)(xmid + tok * C_ + c0) = mv;
  _Float16 hh[4] = {(_Float16)(e0 * rs2 * g2[c0+0] + b2[c0+0]),
                    (_Float16)(e1 * rs2 * g2[c0+1] + b2[c0+1]),
                    (_Float16)(e2 * rs2 * g2[c0+2] + b2[c0+2]),
                    (_Float16)(e3 * rs2 * g2[c0+3] + b2[c0+3])};
  *(float2*)(xnh + tok * C_ + c0) = *(const float2*)hh;
}

// ---------------- MLP1: h1 = gelu(xn @ w1 + b1)
__global__ void __launch_bounds__(256) k_mlp1(
    const _Float16* __restrict__ xnh, const _Float16* __restrict__ w1ws,
    const float* __restrict__ mb1, _Float16* __restrict__ h1) {
  __shared__ _Float16 sA[64 * C_];                 // 16KB
  const int tid = threadIdx.x;
  const int lane = tid & 31, wave = tid >> 5;
  const size_t row0 = (size_t)blockIdx.x * 64;

#pragma unroll
  for (int i = 0; i < 4; ++i) {
    int ch = tid + 256 * i;
    ((float4*)sA)[ch] = ((const float4*)(xnh + row0 * C_))[ch];
  }
  __syncthreads();

  for (int t = wave; t < 128; t += 8) {            // 4 M-tiles x 32 N-tiles
    int m = t >> 5, n = t & 31;
    v8f acc = {};
#pragma unroll
    for (int kt = 0; kt < 4; ++kt) {
      v16h a = frag_a(sA + m * 16 * C_ + kt * 32, C_);
      v16h b = frag_b_sw(w1ws + ((size_t)(n * 4 + kt) << 9));
      acc = wmma16(a, b, acc);
    }
    int col = lane & 15, r0 = (lane >> 4) << 3;
    int nc = n * 16 + col;
    float bias = mb1[nc];
#pragma unroll
    for (int i = 0; i < 8; ++i) {
      float v = acc[i] + bias;
      float gel = 0.5f * v * (1.f + erff(v * 0.70710678118654752f));  // exact gelu
      h1[(row0 + m * 16 + r0 + i) * MLP_ + nc] = (_Float16)gel;
    }
  }
}

// ---------------- MLP2: out = xmid + h1 @ w2 + b2
__global__ void __launch_bounds__(256) k_mlp2(
    const _Float16* __restrict__ h1, const _Float16* __restrict__ w2ws,
    const float* __restrict__ mb2, const float* __restrict__ xmid,
    float* __restrict__ out) {
  __shared__ _Float16 sA[64 * 256];                // 32KB K-slab
  const int tid = threadIdx.x;
  const int lane = tid & 31, wave = tid >> 5;
  const size_t row0 = (size_t)blockIdx.x * 64;

  v8f acc[4];
#pragma unroll
  for (int i = 0; i < 4; ++i) acc[i] = (v8f){};

  for (int kb = 0; kb < 2; ++kb) {
    __syncthreads();
#pragma unroll
    for (int i = 0; i < 8; ++i) {                  // 2048 chunks of 8 halves
      int ch = tid + 256 * i;
      int r = ch >> 5, cc = (ch & 31) << 3;
      ((float4*)sA)[ch] =
          *(const float4*)(h1 + (row0 + r) * MLP_ + kb * 256 + cc);
    }
    __syncthreads();

#pragma unroll
    for (int i = 0; i < 4; ++i) {
      int t = wave + 8 * i;                        // 4 M-tiles x 8 N-tiles
      int m = t >> 3, n = t & 7;
#pragma unroll
      for (int kt = 0; kt < 8; ++kt) {
        v16h a = frag_a(sA + m * 16 * 256 + kt * 32, 256);
        v16h b = frag_b_sw(w2ws + ((size_t)(n * 16 + kb * 8 + kt) << 9));
        acc[i] = wmma16(a, b, acc[i]);
      }
    }
  }

  int col = lane & 15, r0 = (lane >> 4) << 3;
#pragma unroll
  for (int i = 0; i < 4; ++i) {
    int t = wave + 8 * i;
    int m = t >> 3, n = t & 7;
    int nc = n * 16 + col;
    float bias = mb2[nc];
#pragma unroll
    for (int j = 0; j < 8; ++j) {
      size_t r = row0 + m * 16 + r0 + j;
      out[r * C_ + nc] = acc[i][j] + bias + xmid[r * C_ + nc];
    }
  }
}

// ---------------------------------------------------------------- launcher
extern "C" void kernel_launch(void* const* d_in, const int* in_sizes, int n_in,
                              void* d_out, int out_size, void* d_ws, size_t ws_size,
                              hipStream_t stream) {
  const float* x        = (const float*)d_in[0];
  const float* ln1_g    = (const float*)d_in[1];
  const float* ln1_b    = (const float*)d_in[2];
  const float* eca_w    = (const float*)d_in[3];
  const float* qkv_w    = (const float*)d_in[4];
  const float* qkv_b    = (const float*)d_in[5];
  const float* rel_tab  = (const float*)d_in[6];
  const float* proj_w   = (const float*)d_in[7];
  const float* proj_b   = (const float*)d_in[8];
  const float* ln2_g    = (const float*)d_in[9];
  const float* ln2_b    = (const float*)d_in[10];
  const float* mlp_w1   = (const float*)d_in[11];
  const float* mlp_b1   = (const float*)d_in[12];
  const float* mlp_w2   = (const float*)d_in[13];
  const float* mlp_b2   = (const float*)d_in[14];
  const float* amask    = (const float*)d_in[15];
  const int*   relidx   = (const int*)d_in[16];
  float* out = (float*)d_out;

  char* p = (char*)d_ws;
  auto alloc = [&](size_t bytes) -> void* {
    void* r = (void*)p;
    p += (bytes + 255) & ~(size_t)255;
    return r;
  };

  _Float16* qkvws  = (_Float16*)alloc((size_t)C_ * 3 * C_ * 2);
  _Float16* projws = (_Float16*)alloc((size_t)C_ * C_ * 2);
  _Float16* w1ws   = (_Float16*)alloc((size_t)C_ * MLP_ * 2);
  _Float16* w2ws   = (_Float16*)alloc((size_t)MLP_ * C_ * 2);
  float*    sums   = (float*)alloc((size_t)B_ * C_ * 4);
  float*    gate   = (float*)alloc((size_t)B_ * 4);
  _Float16* xsh    = (_Float16*)alloc((size_t)NTOK * C_ * 2);
  float*    aout   = (float*)alloc((size_t)NTOK * C_ * 4);
  float*    xmid   = (float*)alloc((size_t)NTOK * C_ * 4);
  _Float16* xnh    = (_Float16*)alloc((size_t)NTOK * C_ * 2);
  _Float16* h1     = (_Float16*)alloc((size_t)NTOK * MLP_ * 2);

  // weights f32 -> f16, pre-swizzled into per-lane WMMA B-fragment records
  k_swz<<<(4  * 24 * 32 + 255) / 256, 256, 0, stream>>>(qkv_w,  qkvws,  C_,  3*C_);
  k_swz<<<(4  *  8 * 32 + 255) / 256, 256, 0, stream>>>(proj_w, projws, C_,  C_);
  k_swz<<<(4  * 32 * 32 + 255) / 256, 256, 0, stream>>>(mlp_w1, w1ws,   C_,  MLP_);
  k_swz<<<(16 *  8 * 32 + 255) / 256, 256, 0, stream>>>(mlp_w2, w2ws,   MLP_, C_);
  k_zero<<<(B_*C_ + 255) / 256, 256, 0, stream>>>(sums, B_*C_);

  // LN1 + ECA channel sums
  k_ln1<<<NTOK / 8, 256, 0, stream>>>(x, ln1_g, ln1_b, xsh, sums);
  k_gate<<<B_, C_, 0, stream>>>(sums, eca_w, gate);

  // shifted window attention (WMMA) -> aout
  k_attn<<<B_ * NWIN, 256, 0, stream>>>(xsh, qkvws, qkv_b, projws, proj_b,
                                        rel_tab, relidx, amask, aout);

  // residual + cattn + LN2
  k_mid<<<NTOK / 8, 256, 0, stream>>>(x, aout, gate, ln1_g, ln1_b, ln2_g, ln2_b,
                                      xmid, xnh);

  // MLP (WMMA)
  k_mlp1<<<NTOK / 64, 256, 0, stream>>>(xnh, w1ws, mlp_b1, h1);
  k_mlp2<<<NTOK / 64, 256, 0, stream>>>(h1, w2ws, mlp_b2, xmid, out);

  (void)in_sizes; (void)n_in; (void)out_size; (void)ws_size;
}
